// MultiHeadAttention_8091718386486
// MI455X (gfx1250) — matrix operand order repaired and checked by reading, not compiled, not run on hardware
//
#include <hip/hip_runtime.h>

typedef __attribute__((ext_vector_type(16))) __bf16 v16bf;
typedef __attribute__((ext_vector_type(8)))  __bf16 v8bf;
typedef __attribute__((ext_vector_type(8)))  float  v8f;

#define B_  2
#define TQ  2048
#define TV  2048
#define D_  512
#define H_  8
#define DH  64

__device__ __forceinline__ v16bf cat8(v8bf lo, v8bf hi) {
  return __builtin_shufflevector(lo, hi, 0,1,2,3,4,5,6,7,8,9,10,11,12,13,14,15);
}

__device__ __forceinline__ unsigned pack2(float a, float b) {
  union { unsigned u; __bf16 h[2]; } p;
  p.h[0] = (__bf16)a; p.h[1] = (__bf16)b;
  return p.u;
}

// ---- prep: value -> bf16 (row-major) and bf16 transposed [b][d][k] ----
__global__ void prep_value(const float* __restrict__ value,
                           __bf16* __restrict__ vb, __bf16* __restrict__ vt) {
  int idx = blockIdx.x * 256 + threadIdx.x;          // B*TV*DH = 262144
  int c  = idx & 63;
  int kk = (idx >> 6) & 2047;
  int b  = idx >> 17;
  float v = value[idx];
  vb[idx] = (__bf16)v;
  vt[((size_t)(b * DH + c)) * TV + kk] = (__bf16)v;
}

// ---- prep: Qp[b,h,q,in] = sum_c query[b,q,h*64+c] * conv_w[h*64+c, in] ----
__global__ void prep_qp(const float* __restrict__ query,
                        const float* __restrict__ w,
                        __bf16* __restrict__ qp) {
  int idx = blockIdx.x * 256 + threadIdx.x;          // B*H*TQ*64 = 2097152
  int in = idx & 63;
  int q  = (idx >> 6) & 2047;
  int h  = (idx >> 17) & 7;
  int b  = idx >> 20;
  const float* qrow = query + ((size_t)(b * TQ + q)) * D_ + h * DH;
  const float* wcol = w + (size_t)h * DH * DH + in;
  float s = 0.f;
  #pragma unroll 8
  for (int c = 0; c < DH; ++c) s += qrow[c] * wcol[(size_t)c * DH];
  qp[idx] = (__bf16)s;
}

// ---- prep: sbias[b,h,q] = sum_c query[b,q,h*64+c] * conv_b[h*64+c] ----
__global__ void prep_sbias(const float* __restrict__ query,
                           const float* __restrict__ bias,
                           float* __restrict__ sb) {
  int idx = blockIdx.x * 256 + threadIdx.x;          // B*H*TQ = 32768
  int q = idx & 2047;
  int h = (idx >> 11) & 7;
  int b = idx >> 14;
  const float* qrow = query + ((size_t)(b * TQ + q)) * D_ + h * DH;
  const float* brow = bias + h * DH;
  float s = 0.f;
  #pragma unroll 8
  for (int c = 0; c < DH; ++c) s += qrow[c] * brow[c];
  sb[idx] = s;
}

// ---- main fused attention kernel ----
// Grid: B * TQ/16 workgroups, 256 threads (8 waves).
// Wave w: phase A -> scores+softmax(all heads) for key subtile w;
//         phase B -> ctx accumulation for head w.
__launch_bounds__(256, 1)
__global__ void attn_kernel(const __bf16* __restrict__ qp,
                            const __bf16* __restrict__ vb,
                            const __bf16* __restrict__ vt,
                            const float*  __restrict__ sbias,
                            float* __restrict__ out) {
  __shared__ __align__(16) __bf16   QpL[H_][16][DH];     // 16 KB, Qp^T B-operand source
  __shared__ __align__(16) unsigned WL[8][H_][16][8];    // 32 KB, bf16 weight pairs

  const int tid  = threadIdx.x;
  const int wave = tid >> 5;
  const int lane = tid & 31;
  const int lq   = lane & 15;      // N index (query column)
  const int lhi  = lane >> 4;      // lane half

  const int bq = blockIdx.x;
  const int b  = bq >> 7;          // TQ/16 = 128 tiles per batch
  const int q0 = (bq & 127) << 4;

  // stage Qp tile for this (b, q-tile): 8 heads x 16 q x 64 c bf16
  {
    int h = tid >> 5, rem = tid & 31;
    int q = rem >> 1, ch = (rem & 1) * 32;
    const uint4* src = (const uint4*)(qp +
        ((size_t)((b * H_ + h) * TQ) + q0 + q) * DH + ch);
    uint4* dst = (uint4*)(&QpL[h][q][ch]);
    dst[0] = src[0]; dst[1] = src[1]; dst[2] = src[2]; dst[3] = src[3];
  }

  float sb[H_];
  #pragma unroll
  for (int h = 0; h < H_; ++h)
    sb[h] = sbias[(size_t)(b * H_ + h) * TQ + q0 + lq];

  v8f acc[4];
  #pragma unroll
  for (int t = 0; t < 4; ++t) acc[t] = (v8f){0.f,0.f,0.f,0.f,0.f,0.f,0.f,0.f};

  __syncthreads();

  for (int kc = 0; kc < TV; kc += 128) {
    // ---------- phase A: S^T[k,q] for my 16-key subtile, all 8 heads ----------
    const int kb = kc + wave * 16;
    const __bf16* vrow = vb + ((size_t)(b * TV + kb + lq)) * DH;
    // A-operand (16k x 32c): lane holds contiguous c run [lhi*16, +16)
    v16bf A0 = *(const v16bf*)(vrow + lhi * 16);
    v16bf A1 = *(const v16bf*)(vrow + 32 + lhi * 16);

    float e[H_][8];
    float Z[8];
    #pragma unroll
    for (int r = 0; r < 8; ++r) Z[r] = 0.f;

    #pragma unroll
    for (int h = 0; h < H_; ++h) {
      // B-operand (32c x 16q): lane = q, contiguous 16-c run
      v16bf Bq0 = *(const v16bf*)(&QpL[h][lq][lhi * 16]);
      v16bf Bq1 = *(const v16bf*)(&QpL[h][lq][lhi * 16 + 32]);
      v8f c = (v8f){0.f,0.f,0.f,0.f,0.f,0.f,0.f,0.f};
      c = __builtin_amdgcn_wmma_f32_16x16x32_bf16(false, A0, false, Bq0, (short)0, c, false, false);
      c = __builtin_amdgcn_wmma_f32_16x16x32_bf16(false, A1, false, Bq1, (short)0, c, false, false);
      #pragma unroll
      for (int r = 0; r < 8; ++r) {
        float ee = __expf(c[r] * 0.125f + sb[h]);   // scale 1/sqrt(64)
        e[h][r] = ee;
        Z[r] += ee;                                  // softmax over heads: lane-local
      }
    }
    #pragma unroll
    for (int r = 0; r < 8; ++r) Z[r] = __builtin_amdgcn_rcpf(Z[r]);

    // pack normalized weights as bf16 pairs into LDS (B-operand layout for GEMM2)
    #pragma unroll
    for (int h = 0; h < H_; ++h) {
      #pragma unroll
      for (int jj = 0; jj < 4; ++jj) {
        WL[wave][h][lq][lhi * 4 + jj] =
            pack2(e[h][2 * jj] * Z[2 * jj], e[h][2 * jj + 1] * Z[2 * jj + 1]);
      }
    }
    __syncthreads();

    // ---------- phase B: ctx^T[d,q] accumulation for my head (=wave) ----------
    #pragma unroll
    for (int sc = 0; sc < 4; ++sc) {                  // 4 x 32 keys
      v16bf Bw = *(const v16bf*)(&WL[2 * sc + lhi][wave][lq][0]);
      #pragma unroll
      for (int t = 0; t < 4; ++t) {                   // d tiles of 16
        const __bf16* ar = vt + ((size_t)(b * DH + t * 16 + lq)) * TV + kc + sc * 32;
        v16bf Av = *(const v16bf*)(ar + lhi * 16);
        acc[t] = __builtin_amdgcn_wmma_f32_16x16x32_bf16(false, Av, false, Bw, (short)0, acc[t], false, false);
      }
    }
    __syncthreads();
  }

  // ---------- epilogue: out[b, q0+lq, wave*64 + t*16 + lhi*8 + r] ----------
  #pragma unroll
  for (int t = 0; t < 4; ++t) {
    float* o = out + ((size_t)(b * TQ + q0 + lq)) * D_ + wave * DH + t * 16 + lhi * 8;
    float4 f0 = {acc[t][0], acc[t][1], acc[t][2], acc[t][3]};
    float4 f1 = {acc[t][4], acc[t][5], acc[t][6], acc[t][7]};
    *(float4*)(o)     = f0;
    *(float4*)(o + 4) = f1;
  }
}

extern "C" void kernel_launch(void* const* d_in, const int* in_sizes, int n_in,
                              void* d_out, int out_size, void* d_ws, size_t ws_size,
                              hipStream_t stream) {
  const float* query  = (const float*)d_in[0];   // [2,2048,512]
  const float* value  = (const float*)d_in[1];   // [2,2048,64]
  const float* conv_w = (const float*)d_in[2];   // [512,64,1]
  const float* conv_b = (const float*)d_in[3];   // [512]
  float* out = (float*)d_out;                    // [2,2048,512]

  char* ws = (char*)d_ws;
  __bf16* qp = (__bf16*)(ws);                          // 2*8*2048*64 bf16 = 4 MB
  __bf16* vb = (__bf16*)(ws + (4u << 20));             // 2*2048*64 bf16 = 512 KB
  __bf16* vt = (__bf16*)(ws + (4u << 20) + (512u << 10)); // 512 KB
  float*  sb = (float*) (ws + (5u << 20));             // 2*8*2048 f32 = 128 KB

  prep_value<<<(B_ * TV * DH) / 256, 256, 0, stream>>>(value, vb, vt);
  prep_qp   <<<(B_ * H_ * TQ * DH) / 256, 256, 0, stream>>>(query, conv_w, qp);
  prep_sbias<<<(B_ * H_ * TQ) / 256, 256, 0, stream>>>(query, conv_b, sb);
  attn_kernel<<<B_ * (TQ / 16), 256, 0, stream>>>(qp, vb, vt, sb, out);
}
